// LSTM_88716844466217
// MI455X (gfx1250) — compile-verified, matrix operand order checked by reading, not run on hardware
//
#include <hip/hip_runtime.h>

typedef __attribute__((ext_vector_type(16))) _Float16 v16h;
typedef __attribute__((ext_vector_type(8)))  _Float16 v8h;
typedef __attribute__((ext_vector_type(8)))  float    v8f;

#define BSZ   64
#define ILEN  256
#define TLEN  64
#define TTOT  (ILEN + TLEN)      // 320
#define IDIM  512
#define HDIM  1024
#define GDIM  (4 * HDIM)         // 4096 gate columns
#define KTOT  (IDIM + HDIM)      // 1536 fused K
#define KSTEPS (KTOT / 32)       // 48 WMMA K-steps
#define ODIM  128

__device__ __forceinline__ float sigmoidf_(float x) { return 1.0f / (1.0f + __expf(-x)); }

// Build a WMMA 16-half operand from two contiguous 16-byte runs.
__device__ __forceinline__ v16h frag16(const _Float16* __restrict__ p0,
                                       const _Float16* __restrict__ p1) {
    v8h lo = *(const v8h*)p0;
    v8h hi = *(const v8h*)p1;
    v16h r;
#pragma unroll
    for (int i = 0; i < 8; i++) { r[i] = lo[i]; r[i + 8] = hi[i]; }
    return r;
}

// ---------------------------------------------------------------------------
// Pack [W_ih ; W_hh]^T (K=1536 x N=4096) into per-lane WMMA B-fragment order.
// Fragment addressing: ((nt*KSTEPS + ks)*32 + lane)*16 + h
// where per 16x16x32 f16 B layout: n = nt*16 + (lane&15),
// k = ks*32 + (lane>=16 ? 8 : 0) + (h<8 ? h : h+8).
// ---------------------------------------------------------------------------
__global__ void pack_w(const float* __restrict__ w_ih, const float* __restrict__ w_hh,
                       _Float16* __restrict__ fragW) {
    size_t idx = (size_t)blockIdx.x * blockDim.x + threadIdx.x;
    const size_t total = (size_t)(GDIM / 16) * KSTEPS * 32 * 16;
    if (idx >= total) return;
    int h    = (int)(idx & 15);
    int lane = (int)((idx >> 4) & 31);
    size_t rest = idx >> 9;
    int ks = (int)(rest % KSTEPS);
    int nt = (int)(rest / KSTEPS);
    int kbase = (lane >> 4) * 8;
    int k = ks * 32 + kbase + (h < 8 ? h : h + 8);
    int n = nt * 16 + (lane & 15);
    float v = (k < IDIM) ? w_ih[(size_t)n * IDIM + k]
                         : w_hh[(size_t)n * HDIM + (k - IDIM)];
    fragW[idx] = (_Float16)v;
}

// Same packing for the classifier weight (K=1024 x N=128).
__global__ void pack_cw(const float* __restrict__ clf_w, _Float16* __restrict__ fragC) {
    int idx = blockIdx.x * blockDim.x + threadIdx.x;
    const int total = (ODIM / 16) * (HDIM / 32) * 32 * 16;
    if (idx >= total) return;
    int h    = idx & 15;
    int lane = (idx >> 4) & 31;
    int rest = idx >> 9;
    int ks = rest % (HDIM / 32);
    int nt = rest / (HDIM / 32);
    int kbase = (lane >> 4) * 8;
    int k = ks * 32 + kbase + (h < 8 ? h : h + 8);
    int n = nt * 16 + (lane & 15);
    fragC[idx] = (_Float16)clf_w[(size_t)n * HDIM + k];
}

__global__ void cvt_x(const float* __restrict__ inp, _Float16* __restrict__ x16) {
    size_t i = (size_t)blockIdx.x * blockDim.x + threadIdx.x;
    if (i < (size_t)BSZ * ILEN * IDIM) x16[i] = (_Float16)inp[i];
}

__global__ void init_state(_Float16* __restrict__ h0, float* __restrict__ c) {
    int i = blockIdx.x * blockDim.x + threadIdx.x;
    if (i < BSZ * HDIM) { h0[i] = (_Float16)0.0f; c[i] = 0.0f; }
}

// ---------------------------------------------------------------------------
// One LSTM timestep. Grid = 64 blocks (4 batch tiles x 16 hidden tiles of 64).
// Block = 256 threads = 8 waves; wave w computes gate group g = w>>1,
// hidden sub-tiles j = (w&1)*2, (w&1)*2+1. Fused K = [x_t(512) ; h_t(1024)],
// split into two loops so every load is base + constant immediate offset
// (no per-iteration 64-bit address math or cndmask in the WMMA loop).
// ---------------------------------------------------------------------------
__global__ void __launch_bounds__(256) lstm_step(
    int t,
    const _Float16* __restrict__ x16,
    const _Float16* __restrict__ fragW,
    const float*    __restrict__ bias,
    const _Float16* __restrict__ hin,
    _Float16*       __restrict__ hout,
    float*          __restrict__ c,
    _Float16*       __restrict__ hs) {
    __shared__ float gl[4][16][65];   // [gate][m][hidden col], padded vs bank conflicts

    const int tid  = threadIdx.x;
    const int lane = tid & 31;
    const int wave = tid >> 5;
    const int bm0  = (blockIdx.x & 3) * 16;   // batch rows
    const int bn   = blockIdx.x >> 2;         // hidden tile (64 cols)
    const int hc0  = bn * 64;

    const int g   = wave >> 1;                // gate group 0..3 (i,f,g,o)
    const int jp  = (wave & 1) * 2;           // local 16-col sub-tile pair
    const int nt0 = g * 64 + bn * 4 + jp;     // global N-tile index into fragW
    const int nt1 = nt0 + 1;

    const int mrow  = lane & 15;
    const int kbase = (lane >> 4) * 8;
    const int rb    = bm0 + mrow;

    v8f acc0 = {}; v8f acc1 = {};

    // Hoisted bases; inner loops index with compile-time constants so loads
    // fold into the 24-bit IOFFSET of global_load_b128.
    const _Float16* xp = x16 + ((size_t)rb * ILEN + t) * IDIM + kbase;
    const _Float16* hp = hin + (size_t)rb * HDIM + kbase;
    const _Float16* wp0 = fragW + ((size_t)nt0 * KSTEPS * 32 + lane) * 16;
    const _Float16* wp1 = fragW + ((size_t)nt1 * KSTEPS * 32 + lane) * 16;
    // Per K-step the B fragment advances by 32 lanes * 16 halves = 512 halves.

    if (t < ILEN) {                           // x-region K-steps 0..15
#pragma unroll 8
        for (int ks = 0; ks < IDIM / 32; ++ks) {
            v16h A  = frag16(xp  + ks * 32,  xp  + ks * 32 + 16);
            v16h B0 = frag16(wp0 + ks * 512, wp0 + ks * 512 + 8);
            v16h B1 = frag16(wp1 + ks * 512, wp1 + ks * 512 + 8);
            acc0 = __builtin_amdgcn_wmma_f32_16x16x32_f16(false, A, false, B0, (short)0, acc0, false, false);
            acc1 = __builtin_amdgcn_wmma_f32_16x16x32_f16(false, A, false, B1, (short)0, acc1, false, false);
        }
    }
#pragma unroll 8
    for (int ks = IDIM / 32; ks < KSTEPS; ++ks) {   // h-region K-steps 16..47
        const int kh = ks * 32 - IDIM;
        v16h A  = frag16(hp  + kh,       hp  + kh + 16);
        v16h B0 = frag16(wp0 + ks * 512, wp0 + ks * 512 + 8);
        v16h B1 = frag16(wp1 + ks * 512, wp1 + ks * 512 + 8);
        acc0 = __builtin_amdgcn_wmma_f32_16x16x32_f16(false, A, false, B0, (short)0, acc0, false, false);
        acc1 = __builtin_amdgcn_wmma_f32_16x16x32_f16(false, A, false, B1, (short)0, acc1, false, false);
    }

    // Spill accumulators (C/D layout: col = lane&15, row = v + 8*(lane>>4)).
    {
        const int col0 = jp * 16 + (lane & 15);
#pragma unroll
        for (int v = 0; v < 8; v++) {
            const int m = v + 8 * (lane >> 4);
            gl[g][m][col0]      = acc0[v];
            gl[g][m][col0 + 16] = acc1[v];
        }
    }
    __syncthreads();

    // Fused gate nonlinearity + cell/hidden update: 4 (m,col) cells per thread.
    const int col = tid & 63;
    const int mb  = tid >> 6;
#pragma unroll
    for (int r = 0; r < 4; r++) {
        const int m   = mb + 4 * r;
        const int row = bm0 + m;
        const int hc  = hc0 + col;
        float gi = gl[0][m][col] + bias[hc];
        float gf = gl[1][m][col] + bias[HDIM + hc];
        float gg = gl[2][m][col] + bias[2 * HDIM + hc];
        float go = gl[3][m][col] + bias[3 * HDIM + hc];
        float i_ = sigmoidf_(gi);
        float f_ = sigmoidf_(gf);
        float g_ = tanhf(gg);
        float o_ = sigmoidf_(go);
        const size_t ci = (size_t)row * HDIM + hc;
        float cn = f_ * c[ci] + i_ * g_;
        c[ci] = cn;
        float hn = o_ * tanhf(cn);
        hout[ci] = (_Float16)hn;
        if (t >= ILEN) {
            hs[((size_t)row * TLEN + (t - ILEN)) * HDIM + hc] = (_Float16)hn;
        }
    }
}

// ---------------------------------------------------------------------------
// Classifier: probs = sigmoid(hs @ clf_w^T + clf_b). M=4096, N=128, K=1024.
// 256 blocks, 8 waves; wave w owns output N-tile w.
// ---------------------------------------------------------------------------
__global__ void __launch_bounds__(256) classifier(
    const _Float16* __restrict__ hs,
    const _Float16* __restrict__ fragC,
    const float*    __restrict__ clf_b,
    float*          __restrict__ out) {
    const int tid  = threadIdx.x;
    const int lane = tid & 31;
    const int wave = tid >> 5;     // N-tile 0..7
    const int mt   = blockIdx.x;   // M-tile 0..255
    const int kbase = (lane >> 4) * 8;
    const int row  = mt * 16 + (lane & 15);
    const _Float16* ap = hs + (size_t)row * HDIM + kbase;
    const _Float16* bp = fragC + ((size_t)wave * (HDIM / 32) * 32 + lane) * 16;

    v8f acc = {};
#pragma unroll 8
    for (int ks = 0; ks < HDIM / 32; ks++) {
        v16h A = frag16(ap + ks * 32,  ap + ks * 32 + 16);
        v16h B = frag16(bp + ks * 512, bp + ks * 512 + 8);
        acc = __builtin_amdgcn_wmma_f32_16x16x32_f16(false, A, false, B, (short)0, acc, false, false);
    }

    const int o  = wave * 16 + (lane & 15);
    const float bo = clf_b[o];
#pragma unroll
    for (int v = 0; v < 8; v++) {
        const int m = v + 8 * (lane >> 4);
        const int r = mt * 16 + m;
        out[(size_t)r * ODIM + o] = sigmoidf_(acc[v] + bo);
    }
}

// ---------------------------------------------------------------------------
extern "C" void kernel_launch(void* const* d_in, const int* in_sizes, int n_in,
                              void* d_out, int out_size, void* d_ws, size_t ws_size,
                              hipStream_t stream) {
    const float* inp   = (const float*)d_in[0];
    // d_in[1] = tlen (scalar, compile-time constant here)
    const float* w_ih  = (const float*)d_in[2];
    const float* w_hh  = (const float*)d_in[3];
    const float* bias  = (const float*)d_in[4];
    const float* clf_w = (const float*)d_in[5];
    const float* clf_b = (const float*)d_in[6];
    float* out = (float*)d_out;

    char* ws = (char*)d_ws;
    auto alignup = [](size_t x) { return (x + 255) & ~(size_t)255; };
    size_t off = 0;
    _Float16* fragW = (_Float16*)(ws + off);
    off = alignup(off + sizeof(_Float16) * (size_t)(GDIM / 16) * KSTEPS * 32 * 16);   // 12.6 MB
    _Float16* fragC = (_Float16*)(ws + off);
    off = alignup(off + sizeof(_Float16) * (size_t)(ODIM / 16) * (HDIM / 32) * 32 * 16); // 0.26 MB
    _Float16* x16 = (_Float16*)(ws + off);
    off = alignup(off + sizeof(_Float16) * (size_t)BSZ * ILEN * IDIM);                // 16.8 MB
    _Float16* hbuf = (_Float16*)(ws + off);
    off = alignup(off + sizeof(_Float16) * 2 * (size_t)BSZ * HDIM);                   // 0.26 MB
    float* cbuf = (float*)(ws + off);
    off = alignup(off + sizeof(float) * (size_t)BSZ * HDIM);                          // 0.26 MB
    _Float16* hs = (_Float16*)(ws + off);
    off = alignup(off + sizeof(_Float16) * (size_t)BSZ * TLEN * HDIM);                // 8.4 MB

    {
        size_t n = (size_t)(GDIM / 16) * KSTEPS * 32 * 16;
        pack_w<<<(unsigned)((n + 255) / 256), 256, 0, stream>>>(w_ih, w_hh, fragW);
    }
    {
        int n = (ODIM / 16) * (HDIM / 32) * 32 * 16;
        pack_cw<<<(n + 255) / 256, 256, 0, stream>>>(clf_w, fragC);
    }
    {
        size_t n = (size_t)BSZ * ILEN * IDIM;
        cvt_x<<<(unsigned)((n + 255) / 256), 256, 0, stream>>>(inp, x16);
    }
    {
        int n = BSZ * HDIM;
        init_state<<<(n + 255) / 256, 256, 0, stream>>>(hbuf, cbuf);
    }

    for (int t = 0; t < TTOT; t++) {
        const _Float16* hin = hbuf + (size_t)(t & 1) * BSZ * HDIM;
        _Float16* hout      = hbuf + (size_t)((t + 1) & 1) * BSZ * HDIM;
        lstm_step<<<64, 256, 0, stream>>>(t, x16, fragW, bias, hin, hout, cbuf, hs);
    }

    classifier<<<(BSZ * TLEN) / 16, 256, 0, stream>>>(hs, fragC, clf_b, out);

    (void)in_sizes; (void)n_in; (void)out_size; (void)ws_size;
}
